// BurstSnn_42545946034962
// MI455X (gfx1250) — matrix operand order repaired and checked by reading, not compiled, not run on hardware
//
#include <hip/hip_runtime.h>

// ---------------------------------------------------------------------------
// Fused burst-coded SNN scan for MI455X (gfx1250, wave32, WMMA).
//
// Entire 32-step recurrence stays on-chip:
//   - encoder residual r  : LDS  (f32, per-sample x 192 padded features)
//   - encoder threshold   : LDS  (u8 exponent, th = 0.125 * 2^e)
//   - mem1 (16x50 LIF)    : VGPRs, kept in WMMA C/D fragment layout
//   - mem2 (16x5  LIF)    : VGPRs, kept in WMMA C/D fragment layout
// Layer-1 GEMM: spk_in[16x192] @ W1^T -> 4 N-tiles x 6 K-chunks of
//   v_wmma_f32_16x16x32_f16 per wave per step (spikes are exact in f16).
// Layer-2 GEMM: spk1[16x64pad] @ W2^T -> 2 WMMAs via LDS staging.
// ---------------------------------------------------------------------------

typedef __attribute__((ext_vector_type(16))) _Float16 v16h;
typedef __attribute__((ext_vector_type(8)))  float    v8f;

#define BN        16384
#define DFEAT     187
#define KPAD      192      // 6 chunks of 32
#define HID       50
#define NCLS      5
#define TSTEPS    32
#define SPB       32       // samples per block (16 per wave, 2 waves)
#define THREADS   64

__device__ __forceinline__ v8f zero8() {
    v8f v;
#pragma unroll
    for (int i = 0; i < 8; ++i) v[i] = 0.0f;
    return v;
}

__global__ __launch_bounds__(THREADS)
void burst_snn_fused(const float* __restrict__ x,
                     const float* __restrict__ W1,
                     const float* __restrict__ W2,
                     float* __restrict__ out) {
    __shared__ float          sh_r[SPB][KPAD];                 // 24 KB
    __shared__ unsigned char  sh_e[SPB][KPAD];                 //  6 KB
    __shared__ __align__(32) _Float16 sh_B1[6 * 4 * 32 * 16];  // 24 KB, frag-ready
    __shared__ __align__(32) _Float16 sh_spk1[2][16][64];      //  4 KB, per-wave

    const int tid  = threadIdx.x;          // 0..63
    const int wv   = tid >> 5;             // wave id in block
    const int ln   = tid & 31;             // lane in wave (wave32)
    const bool hiL = (ln >= 16);
    const int lid  = ln & 15;              // row (A/sample) or col (B/D) index
    const int sblk = blockIdx.x * SPB;     // first sample of block
    const int gbase = sblk + wv * 16;      // first sample of this wave
    const int sw    = wv * 16;             // wave's sample offset in LDS

    // ---- Phase 1: load x into residual state, init thresholds --------------
    for (int i = tid; i < SPB * KPAD; i += THREADS) {
        const int s = i / KPAD, k = i % KPAD;
        sh_r[s][k] = (k < DFEAT) ? x[(size_t)(sblk + s) * DFEAT + k] : 0.0f;
        sh_e[s][k] = 0;
    }
    // ---- Phase 1b: pre-swizzle W1 into B-fragment layout (f16) -------------
    // B(32x16) frag: lane n holds column n; lanes 16-31 hold K+16; element j->K=j.
    for (int i = tid; i < 6 * 4 * 32 * 16; i += THREADS) {
        const int j    = i & 15;
        const int lane = (i >> 4) & 31;
        const int nb   = (i >> 9) & 3;
        const int cc   = i >> 11;
        const int n    = lane & 15;
        const int h    = nb * 16 + n;
        const int k    = cc * 32 + j + ((lane >= 16) ? 16 : 0);
        _Float16 v = (_Float16)0.0f;
        if (h < HID && k < DFEAT) v = (_Float16)W1[h * DFEAT + k];
        sh_B1[i] = v;
    }
    __syncthreads();

    // ---- Layer-2 B fragments (K=0..31 and K=32..63, padded past 50) --------
    v16h b2a, b2b;
#pragma unroll
    for (int j = 0; j < 16; ++j) {
        const int koff = hiL ? 16 : 0;
        const int k0 = j + koff;
        const int k1 = 32 + j + koff;
        b2a[j] = (lid < NCLS && k0 < HID) ? (_Float16)W2[lid * HID + k0] : (_Float16)0.0f;
        b2b[j] = (lid < NCLS && k1 < HID) ? (_Float16)W2[lid * HID + k1] : (_Float16)0.0f;
    }

    // ---- Recurrent state in registers (C/D fragment layout) ----------------
    v8f mem1[4];
#pragma unroll
    for (int nb = 0; nb < 4; ++nb) mem1[nb] = zero8();
    v8f mem2 = zero8();
    float cnt = 0.0f;

    const size_t cnt_off = (size_t)TSTEPS * BN * NCLS;

    for (int t = 0; t < TSTEPS; ++t) {
        // ---------- layer 1: encoder fused into A-frag build + 24 WMMAs -----
        v8f cur1[4];
#pragma unroll
        for (int nb = 0; nb < 4; ++nb) cur1[nb] = zero8();

        for (int cc = 0; cc < 6; ++cc) {
            // A(16x32) frag: lane l row l; element j -> K = j (+16 hi)
            v16h a;
            const int s = sw + lid;
#pragma unroll
            for (int j = 0; j < 16; ++j) {
                const int k = cc * 32 + j + (hiL ? 16 : 0);
                float r = sh_r[s][k];
                int   e = sh_e[s][k];
                const float th = 0.125f * (float)(1 << e);
                const bool sp = (r >= th);
                sh_r[s][k] = sp ? (r - th) : r;
                sh_e[s][k] = sp ? (unsigned char)(e + 1) : (unsigned char)0;
                a[j] = sp ? (_Float16)1.0f : (_Float16)0.0f;
                cnt += sp ? 1.0f : 0.0f;
            }
#pragma unroll
            for (int nb = 0; nb < 4; ++nb) {
                const v16h b = *(const v16h*)&sh_B1[((cc * 4 + nb) * 32 + ln) * 16];
                cur1[nb] = __builtin_amdgcn_wmma_f32_16x16x32_f16(
                    false, a, false, b, (short)0, cur1[nb], false, false);
            }
        }

        // ---------- LIF layer 1 + stage spikes into LDS ---------------------
#pragma unroll
        for (int nb = 0; nb < 4; ++nb) {
            const int h = nb * 16 + lid;
#pragma unroll
            for (int rr = 0; rr < 8; ++rr) {
                float v = 0.9f * mem1[nb][rr] + cur1[nb][rr];
                const float sp = (v > 1.0f) ? 1.0f : 0.0f;
                v -= sp;
                mem1[nb][rr] = v;
                const int mrow = rr + (hiL ? 8 : 0);
                sh_spk1[wv][mrow][h] = (h < HID) ? (_Float16)sp : (_Float16)0.0f;
            }
        }
        __syncthreads();   // uniform; makes cross-lane LDS staging visible

        // ---------- layer 2: 2 WMMAs over K=64 (padded 50) ------------------
        v8f cur2 = zero8();
        {
            v16h a2;
#pragma unroll
            for (int j = 0; j < 16; ++j) {
                const int k2 = j + (hiL ? 16 : 0);
                a2[j] = sh_spk1[wv][lid][k2];
            }
            cur2 = __builtin_amdgcn_wmma_f32_16x16x32_f16(
                false, a2, false, b2a, (short)0, cur2, false, false);
#pragma unroll
            for (int j = 0; j < 16; ++j) {
                const int k2 = 32 + j + (hiL ? 16 : 0);
                a2[j] = sh_spk1[wv][lid][k2];
            }
            cur2 = __builtin_amdgcn_wmma_f32_16x16x32_f16(
                false, a2, false, b2b, (short)0, cur2, false, false);
        }

        // ---------- LIF layer 2 + spike record ------------------------------
#pragma unroll
        for (int rr = 0; rr < 8; ++rr) {
            float v = 0.9f * mem2[rr] + cur2[rr];
            const float sp = (v > 1.0f) ? 1.0f : 0.0f;
            v -= sp;
            mem2[rr] = v;
            if (lid < NCLS) {
                const int mrow = rr + (hiL ? 8 : 0);
                out[(size_t)t * BN * NCLS + (size_t)(gbase + mrow) * NCLS + lid] = sp;
            }
        }
    }

    // ---- spike counts: lanes l and l+16 hold complementary K halves --------
    const float tot = cnt + __shfl_xor(cnt, 16, 32);
    if (!hiL) out[cnt_off + gbase + lid] = tot;
}

extern "C" void kernel_launch(void* const* d_in, const int* in_sizes, int n_in,
                              void* d_out, int out_size, void* d_ws, size_t ws_size,
                              hipStream_t stream) {
    const float* x  = (const float*)d_in[0];   // [16384, 187]
    const float* W1 = (const float*)d_in[1];   // [50, 187]
    const float* W2 = (const float*)d_in[2];   // [5, 50]
    float* out = (float*)d_out;                // [32*16384*5] spk_rec ++ [16384] counts

    dim3 grid(BN / SPB);   // 512 blocks, 32 samples each
    dim3 block(THREADS);   // 2 waves/block, 16 samples/wave
    burst_snn_fused<<<grid, block, 0, stream>>>(x, W1, W2, out);
}
